// SACapsuleCONV_79817672228991
// MI455X (gfx1250) — compile-verified, matrix operand order checked by reading, not run on hardware
//
#include <hip/hip_runtime.h>

typedef __attribute__((ext_vector_type(16))) _Float16 v16h;
typedef __attribute__((ext_vector_type(8)))  float    v8f;

#define KK    3
#define DD    16
#define IN_N  32
#define OUT_N 32
#define HH    28
#define WW    28
#define HO    26
#define WO    26
#define II    288            // IN_N * KK * KK
#define ITERS 7
#define EPSV  1e-5f
#define RSTR  36             // padded row stride for r (16B aligned, conflict-free cols)

__global__ __launch_bounds__(256) void caps_kernel(
    const float* __restrict__ inp,   // (4,32,28,28,16)
    const float* __restrict__ wc,    // (3,3,32,4,4)
    const float* __restrict__ wn,    // (32,4,4) -> qf (32,16)
    const float* __restrict__ lns,   // (16)
    const float* __restrict__ lnb,   // (16)
    float* __restrict__ out)         // (4,32,26,26,16)
{
    __shared__ float sV[II][DD];        // Vf, 18 KB
    __shared__ float sR[II][RSTR];      // r / attn, 40.5 KB
    __shared__ float sRed[OUT_N][8];    // column-reduction scratch
    __shared__ float sP[OUT_N][DD];     // next_pose

    const int blk  = blockIdx.x;
    const int b    = blk / (HO * WO);
    const int rem  = blk % (HO * WO);
    const int h0   = rem / WO;
    const int w0   = rem % WO;
    const int t    = threadIdx.x;
    const int lane = t & 31;
    const int wave = t >> 5;

    // ---- Step 1: V[i] = pose(b, n, h0+k, w0+l) @ w_current[k,l,n]  (4x4 each) ----
    for (int i = t; i < II; i += 256) {
        const int n = i / 9, kl = i % 9, k = kl / 3, l = kl % 3;
        const float* P  = inp + (((size_t)(b * IN_N + n) * HH + (h0 + k)) * WW + (w0 + l)) * DD;
        const float* Wm = wc  + ((k * KK + l) * IN_N + n) * DD;
        float p[16];
        #pragma unroll
        for (int j = 0; j < 16; ++j) p[j] = P[j];
        #pragma unroll
        for (int pr = 0; pr < 4; ++pr)
            #pragma unroll
            for (int r = 0; r < 4; ++r) {
                float acc = 0.f;
                #pragma unroll
                for (int q = 0; q < 4; ++q) acc += p[pr * 4 + q] * Wm[q * 4 + r];
                sV[i][pr * 4 + r] = acc;
            }
    }
    __syncthreads();

    // ---- Step 2: dots = Vf(288x16) @ qf^T(16x32) / 3 via WMMA (K=16 padded to 32) ----
    // 18 M-tiles x 2 N-tiles = 36 WMMAs, distributed over 8 waves.
    for (int idx = wave; idx < 36; idx += 8) {
        const int m  = idx >> 1;
        const int nt = idx & 1;
        const int khalf = (lane >= 16) ? 8 : 0;   // lanes 16-31 hold K=8..15 (and 24..31)
        const int row   = m * 16 + (lane & 15);
        v16h a, bm;
        #pragma unroll
        for (int j = 0; j < 8; ++j) {
            a[j]     = (_Float16)sV[row][khalf + j];   // K = khalf+j  (real, d index)
            a[j + 8] = (_Float16)0.0f;                 // K = 16.. pad
        }
        const int o = nt * 16 + (lane & 15);
        #pragma unroll
        for (int j = 0; j < 8; ++j) {
            bm[j]     = (_Float16)wn[o * 16 + khalf + j];  // B[k][o] = qf[o][k]
            bm[j + 8] = (_Float16)0.0f;
        }
        v8f c = {};
        c = __builtin_amdgcn_wmma_f32_16x16x32_f16(false, a, false, bm, (short)0, c, false, false);
        const int rbase = m * 16 + ((lane >= 16) ? 8 : 0);
        const int ocol  = nt * 16 + (lane & 15);
        #pragma unroll
        for (int v = 0; v < 8; ++v)
            sR[rbase + v][ocol] = c[v] * (1.0f / 3.0f);   // /sqrt(16)/TEMP
    }
    __syncthreads();

    // ---- Step 3: Sinkhorn (7 iters of row- then column-logsumexp) ----
    for (int it = 0; it < ITERS; ++it) {
        // axis -1 (over o=0..31): one row per thread
        for (int i = t; i < II; i += 256) {
            float mx = -3.4e38f;
            #pragma unroll
            for (int o = 0; o < 32; ++o) mx = fmaxf(mx, sR[i][o]);
            float s = 0.f;
            #pragma unroll
            for (int o = 0; o < 32; ++o) s += __expf(sR[i][o] - mx);
            const float lse = mx + __logf(s);
            #pragma unroll
            for (int o = 0; o < 32; ++o) sR[i][o] -= lse;
        }
        __syncthreads();
        // axis -2 (over i=0..287): 8 threads per column
        {
            const int col = t >> 3, sub = t & 7;
            float pm = -3.4e38f;
            for (int i = sub; i < II; i += 8) pm = fmaxf(pm, sR[i][col]);
            sRed[col][sub] = pm;
            __syncthreads();
            float mx = sRed[col][0];
            #pragma unroll
            for (int j = 1; j < 8; ++j) mx = fmaxf(mx, sRed[col][j]);
            __syncthreads();                 // all reads of sRed done before rewrite
            float ps = 0.f;
            for (int i = sub; i < II; i += 8) ps += __expf(sR[i][col] - mx);
            sRed[col][sub] = ps;
            __syncthreads();
            float s = 0.f;
            #pragma unroll
            for (int j = 0; j < 8; ++j) s += sRed[col][j];
            const float lse = mx + __logf(s);
            for (int i = sub; i < II; i += 8) sR[i][col] -= lse;
        }
        __syncthreads();
    }
    // attn = exp(r)
    for (int e = t; e < II * 32; e += 256) {
        const int i = e >> 5, o = e & 31;
        sR[i][o] = __expf(sR[i][o]);
    }
    __syncthreads();

    // ---- Step 4: next_pose = attn^T(32x288) @ Vf(288x16), 2 o-tiles x 9 chained WMMAs ----
    if (wave < 2) {
        const int ot = wave;
        const int khalf = (lane >= 16) ? 8 : 0;
        const int orow  = ot * 16 + (lane & 15);  // A row (o index)
        const int dcol  = lane & 15;              // B col (d index)
        v8f c = {};
        for (int kb = 0; kb < II; kb += 32) {
            v16h a, bm;
            #pragma unroll
            for (int j = 0; j < 8; ++j) {
                a[j]      = (_Float16)sR[kb + khalf + j][orow];        // K = khalf+j
                a[j + 8]  = (_Float16)sR[kb + 16 + khalf + j][orow];   // K = 16+khalf+j
                bm[j]     = (_Float16)sV[kb + khalf + j][dcol];
                bm[j + 8] = (_Float16)sV[kb + 16 + khalf + j][dcol];
            }
            c = __builtin_amdgcn_wmma_f32_16x16x32_f16(false, a, false, bm, (short)0, c, false, false);
        }
        const int obase = ot * 16 + ((lane >= 16) ? 8 : 0);
        #pragma unroll
        for (int v = 0; v < 8; ++v)
            sP[obase + v][lane & 15] = c[v];
    }
    __syncthreads();

    // ---- Step 5: out[o] = next_pose[o](4x4) @ wn[o](4x4), then LayerNorm over 16 ----
    if (t < OUT_N) {
        const int o = t;
        float res[16];
        #pragma unroll
        for (int p = 0; p < 4; ++p)
            #pragma unroll
            for (int r = 0; r < 4; ++r) {
                float acc = 0.f;
                #pragma unroll
                for (int q = 0; q < 4; ++q) acc += sP[o][p * 4 + q] * wn[o * 16 + q * 4 + r];
                res[p * 4 + r] = acc;
            }
        float mu = 0.f;
        #pragma unroll
        for (int d = 0; d < 16; ++d) mu += res[d];
        mu *= (1.0f / 16.0f);
        float var = 0.f;
        #pragma unroll
        for (int d = 0; d < 16; ++d) { const float df = res[d] - mu; var += df * df; }
        var *= (1.0f / 16.0f);
        const float inv = rsqrtf(var + EPSV);
        float* op = out + (((size_t)(b * OUT_N + o) * HO + h0) * WO + w0) * DD;
        #pragma unroll
        for (int d = 0; d < 16; ++d)
            op[d] = (res[d] - mu) * inv * lns[d] + lnb[d];
    }
}

extern "C" void kernel_launch(void* const* d_in, const int* in_sizes, int n_in,
                              void* d_out, int out_size, void* d_ws, size_t ws_size,
                              hipStream_t stream) {
    const float* inp = (const float*)d_in[0];   // input
    const float* wc  = (const float*)d_in[1];   // w_current
    const float* wn  = (const float*)d_in[2];   // w_next
    const float* lns = (const float*)d_in[3];   // ln_scale
    const float* lnb = (const float*)d_in[4];   // ln_bias
    float* outp = (float*)d_out;

    dim3 grid(4 * HO * WO);   // one workgroup per (b, h, w) site
    dim3 block(256);          // 8 wave32 waves
    caps_kernel<<<grid, block, 0, stream>>>(inp, wc, wn, lns, lnb, outp);
}
